// SimpleRNN_14173392077107
// MI455X (gfx1250) — compile-verified, matrix operand order checked by reading, not run on hardware
//
#include <hip/hip_runtime.h>

// ---------------- problem constants ----------------
constexpr int kSamples = 32;
constexpr int kSteps   = 128;
constexpr int kD       = 64;        // input size
constexpr int kO       = 64;        // output size
constexpr int kH       = 1024;      // hidden
constexpr int kG       = 4 * kH;    // 4096 gate rows
constexpr int kT       = kSteps * kSamples;   // 4096 flattened time steps

// persistent scan kernel geometry
constexpr int kNB  = 128;           // persistent blocks (grid-sync participants)
constexpr int kHB  = 8;             // hidden units owned per block (kNB*kHB == kH)
constexpr int kTPR = 4;             // threads per gate-row (64 rows * 4 = 256 threads)
constexpr int kKCH = kH / kTPR;     // 256 K-elements per thread

// LDS weight stash: one 512B chunk per (row,sub); L1 rows have two (Wih1, Whh1).
constexpr int kChunkB      = kKCH * 2;        // 512 bytes of bf16 weights
constexpr int kChunkStride = kChunkB + 16;    // 528B: 16B-aligned, spreads LDS banks
constexpr int kNumChunks   = 128 * 1 + 128 * 2;               // 384 chunks
constexpr int kDynLds      = kNumChunks * kChunkStride;       // ~198 KB (< 320KB WGP LDS)

// two-level barrier geometry
constexpr int kGroups   = 16;
constexpr int kPerGroup = kNB / kGroups;      // 8

typedef __attribute__((ext_vector_type(16))) __bf16 v16bf;
typedef __attribute__((ext_vector_type(8)))  float  v8f;

// ---------------- small helpers ----------------
__device__ __forceinline__ float bflo(unsigned w) {            // low bf16 of a dword -> f32
  return __builtin_bit_cast(float, (unsigned)(w << 16));
}
__device__ __forceinline__ float bfhi(unsigned w) {            // high bf16 of a dword -> f32
  return __builtin_bit_cast(float, (unsigned)(w & 0xFFFF0000u));
}
__device__ __forceinline__ unsigned short f2bf(float f) {      // f32 -> bf16 (RNE)
  unsigned u = __builtin_bit_cast(unsigned, f);
  u += 0x7FFFu + ((u >> 16) & 1u);
  return (unsigned short)(u >> 16);
}
__device__ __forceinline__ float sigf(float x) { return 1.0f / (1.0f + __expf(-x)); }

__device__ __forceinline__ v8f wmma_bf16(v16bf a, v16bf b, v8f c) {
  return __builtin_amdgcn_wmma_f32_16x16x32_bf16(false, a, false, b, (short)0, c, false, false);
}

// bf16-weight dot (kKCH elements) against an LDS-resident fp32 vector chunk.
// `w` may point into LDS (ds_load_b128) or global memory.
__device__ __forceinline__ float dot_bf16(const unsigned short* __restrict__ w,
                                          const float* __restrict__ h) {
  float p = 0.0f;
  #pragma unroll 4
  for (int k = 0; k < kKCH; k += 8) {
    const uint4 wv = *(const uint4*)(const void*)(w + k);
    p = fmaf(bflo(wv.x), h[k + 0], p); p = fmaf(bfhi(wv.x), h[k + 1], p);
    p = fmaf(bflo(wv.y), h[k + 2], p); p = fmaf(bfhi(wv.y), h[k + 3], p);
    p = fmaf(bflo(wv.z), h[k + 4], p); p = fmaf(bfhi(wv.z), h[k + 5], p);
    p = fmaf(bflo(wv.w), h[k + 6], p); p = fmaf(bfhi(wv.w), h[k + 7], p);
  }
  return p;
}

// ---------------- utility kernels ----------------
__global__ void cvt_f32_bf16(const float* __restrict__ in, unsigned short* __restrict__ out, int n) {
  int i = blockIdx.x * blockDim.x + threadIdx.x;
  if (i < n) out[i] = f2bf(in[i]);
}

__global__ void zero_f32(float* __restrict__ p, int n) {
  int i = blockIdx.x * blockDim.x + threadIdx.x;
  if (i < n) p[i] = 0.0f;
}

__global__ void zero_u32(unsigned* __restrict__ p, int n) {
  int i = blockIdx.x * blockDim.x + threadIdx.x;
  if (i < n) p[i] = 0u;
}

// z[t,h] = b_in[h] + dot(x_t, W_in[h,:]) ; stored bf16 for the WMMA GEMM.
// t = step*kSamples + s, and x_t lives at inputs[(s*kSteps + step)*kD ...]
__global__ void z_kernel(const float* __restrict__ inputs, const float* __restrict__ W_in,
                         const float* __restrict__ b_in, unsigned short* __restrict__ zbf) {
  int idx = blockIdx.x * blockDim.x + threadIdx.x;
  if (idx >= kT * kH) return;
  int t = idx / kH, h = idx - t * kH;
  int s = t & (kSamples - 1), step = t >> 5;
  const float* xrow = inputs + ((size_t)s * kSteps + step) * kD;
  const float* wrow = W_in + (size_t)h * kD;
  float acc = b_in[h];
  #pragma unroll 8
  for (int d = 0; d < kD; ++d) acc = fmaf(xrow[d], wrow[d], acc);
  zbf[idx] = f2bf(acc);
}

// ---------------- WMMA GEMM: pre0 = z @ W_ih0^T + (b_ih0 + b_hh0) ----------------
// Register-blocked: one wave computes a 16x64 strip (1 M-tile x 4 N-tiles).
// A fragment loaded once per K-chunk, 4 independent accumulators -> no
// WMMA->WMMA RAW hazard stalls; 2.5 B128 loads per WMMA.
__global__ void gemm_pre0(const unsigned short* __restrict__ Zbf,
                          const unsigned short* __restrict__ Wbf,
                          const float* __restrict__ b_ih0, const float* __restrict__ b_hh0,
                          float* __restrict__ pre0) {
  const int wave = threadIdx.x >> 5;
  const int lane = threadIdx.x & 31;
  const int job  = blockIdx.x * 8 + wave;     // (kT/16) * (kG/64) jobs
  const int NT4  = kG / 64;                   // 64 strip columns
  const int tm   = job / NT4, tn4 = job - tm * NT4;
  const int l15  = lane & 15;
  const int kh   = (lane >> 4) * 16;          // K-half select: k = 16h + s for both A and B (contiguous)
  const unsigned short* arow = Zbf + (size_t)(tm * 16 + l15) * kH;
  const unsigned short* b0 = Wbf + (size_t)(tn4 * 64 + l15) * kH;
  const unsigned short* b1 = b0 + (size_t)16 * kH;
  const unsigned short* b2 = b0 + (size_t)32 * kH;
  const unsigned short* b3 = b0 + (size_t)48 * kH;
  v8f acc0 = {}, acc1 = {}, acc2 = {}, acc3 = {};
  #pragma unroll 2
  for (int kk = 0; kk < kH; kk += 32) {
    const v16bf a   = *(const v16bf*)(const void*)(arow + kk + kh);
    const v16bf bb0 = *(const v16bf*)(const void*)(b0 + kk + kh);
    const v16bf bb1 = *(const v16bf*)(const void*)(b1 + kk + kh);
    const v16bf bb2 = *(const v16bf*)(const void*)(b2 + kk + kh);
    const v16bf bb3 = *(const v16bf*)(const void*)(b3 + kk + kh);
    acc0 = wmma_bf16(a, bb0, acc0);
    acc1 = wmma_bf16(a, bb1, acc1);
    acc2 = wmma_bf16(a, bb2, acc2);
    acc3 = wmma_bf16(a, bb3, acc3);
  }
  const int mb = tm * 16 + 8 * (lane >> 4);   // C/D layout: lanes 16-31 hold M=r+8
  #pragma unroll
  for (int nt = 0; nt < 4; ++nt) {
    const int n = tn4 * 64 + nt * 16 + l15;
    const float bias = b_ih0[n] + b_hh0[n];
    const v8f acc = (nt == 0) ? acc0 : (nt == 1) ? acc1 : (nt == 2) ? acc2 : acc3;
    #pragma unroll
    for (int r = 0; r < 8; ++r) {
      pre0[(size_t)(mb + r) * kG + n] = acc[r] + bias;
    }
  }
}

// ---------------- WMMA GEMM: out = hs @ W_out^T + b_out, fused (t -> s,step) transpose --------
// One wave computes a full 16x64 output strip (all 4 N-tiles of one M-tile).
__global__ void gemm_out(const unsigned short* __restrict__ hsbf,
                         const unsigned short* __restrict__ woutbf,
                         const float* __restrict__ b_out, float* __restrict__ out) {
  const int wave = threadIdx.x >> 5;
  const int lane = threadIdx.x & 31;
  const int tm   = blockIdx.x * 8 + wave;     // kT/16 = 256 jobs
  const int l15  = lane & 15;
  const int kh   = (lane >> 4) * 16;
  const unsigned short* arow = hsbf + (size_t)(tm * 16 + l15) * kH;
  const unsigned short* b0 = woutbf + (size_t)l15 * kH;
  const unsigned short* b1 = b0 + (size_t)16 * kH;
  const unsigned short* b2 = b0 + (size_t)32 * kH;
  const unsigned short* b3 = b0 + (size_t)48 * kH;
  v8f acc0 = {}, acc1 = {}, acc2 = {}, acc3 = {};
  #pragma unroll 2
  for (int kk = 0; kk < kH; kk += 32) {
    const v16bf a   = *(const v16bf*)(const void*)(arow + kk + kh);
    const v16bf bb0 = *(const v16bf*)(const void*)(b0 + kk + kh);
    const v16bf bb1 = *(const v16bf*)(const void*)(b1 + kk + kh);
    const v16bf bb2 = *(const v16bf*)(const void*)(b2 + kk + kh);
    const v16bf bb3 = *(const v16bf*)(const void*)(b3 + kk + kh);
    acc0 = wmma_bf16(a, bb0, acc0);
    acc1 = wmma_bf16(a, bb1, acc1);
    acc2 = wmma_bf16(a, bb2, acc2);
    acc3 = wmma_bf16(a, bb3, acc3);
  }
  const int mb = tm * 16 + 8 * (lane >> 4);
  #pragma unroll
  for (int nt = 0; nt < 4; ++nt) {
    const int o = nt * 16 + l15;
    const float bias = b_out[o];
    const v8f acc = (nt == 0) ? acc0 : (nt == 1) ? acc1 : (nt == 2) ? acc2 : acc3;
    #pragma unroll
    for (int r = 0; r < 8; ++r) {
      const int t = mb + r;
      const int s = t & (kSamples - 1);
      const int step = t >> 5;
      out[((size_t)s * kSteps + step) * kO + o] = acc[r] + bias;
    }
  }
}

// ---------------- two-level grid barrier ----------------
// syncv layout (dwords): leaf counter g at [g*32] (one per 128B line),
// root counter at [520], sense at [552]. Serialized same-line atomics drop
// from kNB to ~kPerGroup + kGroups.
__device__ __forceinline__ void grid_barrier2(unsigned* __restrict__ syncv, int grp) {
  __syncthreads();
  if (threadIdx.x == 0) {
    __threadfence();
    unsigned* leaf  = syncv + grp * 32;
    unsigned* root  = syncv + 520;
    unsigned* sense = syncv + 552;
    const unsigned my = __hip_atomic_load(sense, __ATOMIC_ACQUIRE, __HIP_MEMORY_SCOPE_AGENT);
    const unsigned lp = __hip_atomic_fetch_add(leaf, 1u, __ATOMIC_ACQ_REL, __HIP_MEMORY_SCOPE_AGENT);
    if (lp == (unsigned)(kPerGroup - 1)) {
      __hip_atomic_store(leaf, 0u, __ATOMIC_RELAXED, __HIP_MEMORY_SCOPE_AGENT);
      const unsigned rp = __hip_atomic_fetch_add(root, 1u, __ATOMIC_ACQ_REL, __HIP_MEMORY_SCOPE_AGENT);
      if (rp == (unsigned)(kGroups - 1)) {
        __hip_atomic_store(root, 0u, __ATOMIC_RELAXED, __HIP_MEMORY_SCOPE_AGENT);
        __hip_atomic_fetch_add(sense, 1u, __ATOMIC_RELEASE, __HIP_MEMORY_SCOPE_AGENT);
      } else {
        while (__hip_atomic_load(sense, __ATOMIC_ACQUIRE, __HIP_MEMORY_SCOPE_AGENT) == my) {
          __builtin_amdgcn_s_sleep(2);
        }
      }
    } else {
      while (__hip_atomic_load(sense, __ATOMIC_ACQUIRE, __HIP_MEMORY_SCOPE_AGENT) == my) {
        __builtin_amdgcn_s_sleep(2);
      }
    }
  }
  __syncthreads();
}

// ---------------- persistent LSTM scan (pipelined, LDS-resident weights) ----------------
// Phase t (t = 0..kT) computes layer-0 of step t AND layer-1 of step t-1
// concurrently -> ONE grid barrier per phase (4097 total).
// Block b owns hidden units [b*kHB, (b+1)*kHB): rows 0..31 = layer-0 gates,
// rows 32..63 = layer-1 gates; 4 threads per row over K-chunks of 256.
// Each block's 192KB weight slice (Whh0/Wih1/Whh1 rows) is staged into the
// WGP's LDS ONCE, so the 4096-phase inner loop reads weights at LDS bandwidth
// with zero per-phase L2 traffic.
// Slot convention: h0(s) and h1(s) live in double-buffer slot (s+1)&1.
__launch_bounds__(256, 1)
__global__ void lstm_scan(const float* __restrict__ pre0,
                          const unsigned short* __restrict__ Whh0,
                          const unsigned short* __restrict__ Wih1,
                          const unsigned short* __restrict__ Whh1,
                          const float* __restrict__ b_ih1,
                          const float* __restrict__ b_hh1,
                          float* __restrict__ h0buf,    // 2*kH (double-buffered)
                          float* __restrict__ c0,       // kH
                          float* __restrict__ h1buf,    // 2*kH (double-buffered)
                          float* __restrict__ c1,       // kH
                          unsigned short* __restrict__ hs_bf,  // kT*kH
                          unsigned* __restrict__ syncv) {
  extern __shared__ unsigned char smem_dyn[];   // kDynLds bytes of weight stash
  __shared__ float sh0[kH];        // h0(t-1)
  __shared__ float sh1[kH];        // h1(t-2)
  __shared__ float red[64][kTPR];
  __shared__ float gv0[4][kHB];
  __shared__ float gv1[4][kHB];

  const int tid  = threadIdx.x;
  const int row  = tid >> 2;          // 0..63
  const int sub  = tid & (kTPR - 1);  // 0..3
  const bool isL1 = row >= 32;
  const int rl   = row & 31;          // row within layer
  const int u    = rl >> 2;           // hidden unit 0..7
  const int gate = rl & 3;            // 0=i 1=f 2=g 3=o
  const int hidx = blockIdx.x * kHB + u;
  const int j    = gate * kH + hidx;  // gate row in [0, 4096)
  const int grp  = blockIdx.x >> 3;   // barrier leaf group

  // ---- stage this thread's weight chunk(s) into LDS (once) ----
  const unsigned short* g0  = Whh0 + (size_t)j * kH + sub * kKCH;
  const unsigned short* g1i = Wih1 + (size_t)j * kH + sub * kKCH;
  const unsigned short* g1h = Whh1 + (size_t)j * kH + sub * kKCH;
  unsigned short* myw0  = nullptr;
  unsigned short* myw1i = nullptr;
  unsigned short* myw1h = nullptr;
  if (!isL1) {
    myw0 = (unsigned short*)(void*)(smem_dyn + (size_t)tid * kChunkStride);
    uint4* dst = (uint4*)(void*)myw0;
    const uint4* src = (const uint4*)(const void*)g0;
    #pragma unroll 4
    for (int i = 0; i < kChunkB / 16; ++i) dst[i] = src[i];
  } else {
    const int base = 128 + 2 * (tid - 128);
    myw1i = (unsigned short*)(void*)(smem_dyn + (size_t)base * kChunkStride);
    myw1h = (unsigned short*)(void*)(smem_dyn + (size_t)(base + 1) * kChunkStride);
    uint4* di = (uint4*)(void*)myw1i;
    uint4* dh = (uint4*)(void*)myw1h;
    const uint4* si = (const uint4*)(const void*)g1i;
    const uint4* sh = (const uint4*)(const void*)g1h;
    #pragma unroll 4
    for (int i = 0; i < kChunkB / 16; ++i) { di[i] = si[i]; dh[i] = sh[i]; }
  }
  __syncthreads();

  for (int t = 0; t <= kT; ++t) {
    // h0(t-1) is in slot t&1; h0(t) goes to slot (t+1)&1.
    const float* h0r = h0buf + ((t & 1) ? kH : 0);
    float*       h0w = h0buf + ((t & 1) ? 0 : kH);
    // h1(t-2) is in slot (t-1)&1 == (t+1)&1; h1(t-1) goes to slot t&1.
    const float* h1r = h1buf + ((t & 1) ? 0 : kH);
    float*       h1w = h1buf + ((t & 1) ? kH : 0);

    for (int i = tid; i < kH; i += 256) { sh0[i] = h0r[i]; sh1[i] = h1r[i]; }
    __syncthreads();

    float p = 0.0f;
    if (!isL1) {
      // layer-0 matvec for step t: Whh0 @ h0(t-1)     (LDS-resident weights)
      if (t < kT) p = dot_bf16(myw0, sh0 + sub * kKCH);
    } else {
      // layer-1 matvec for step t-1: Wih1 @ h0(t-1) + Whh1 @ h1(t-2)
      if (t >= 1) p = dot_bf16(myw1i, sh0 + sub * kKCH) + dot_bf16(myw1h, sh1 + sub * kKCH);
    }
    red[row][sub] = p;
    __syncthreads();

    if (tid < 64) {
      float s = red[tid][0];
      #pragma unroll
      for (int q = 1; q < kTPR; ++q) s += red[tid][q];
      if (tid < 32) {
        const int jg = (tid & 3) * kH + blockIdx.x * kHB + (tid >> 2);
        float v = s;
        if (t < kT) {
          v += pre0[(size_t)t * kG + jg];   // pre0 already contains b_ih0+b_hh0
          if (t + 1 < kT) __builtin_prefetch(pre0 + (size_t)(t + 1) * kG + jg, 0, 1);
        }
        gv0[tid & 3][tid >> 2] = v;
      } else {
        const int r = tid - 32;
        const int jg = (r & 3) * kH + blockIdx.x * kHB + (r >> 2);
        gv1[r & 3][r >> 2] = s + b_ih1[jg] + b_hh1[jg];
      }
    }
    __syncthreads();

    if (tid < kHB) {
      // layer-0 state update for step t
      if (t < kT) {
        const int idx = blockIdx.x * kHB + tid;
        const float iv = sigf(gv0[0][tid]);
        const float fv = sigf(gv0[1][tid]);
        const float gg = tanhf(gv0[2][tid]);
        const float ov = sigf(gv0[3][tid]);
        const float c  = fv * c0[idx] + iv * gg;
        c0[idx]  = c;
        h0w[idx] = ov * tanhf(c);
      }
    } else if (tid < 2 * kHB) {
      // layer-1 state update for step t-1
      if (t >= 1) {
        const int uu  = tid - kHB;
        const int idx = blockIdx.x * kHB + uu;
        const float iv = sigf(gv1[0][uu]);
        const float fv = sigf(gv1[1][uu]);
        const float gg = tanhf(gv1[2][uu]);
        const float ov = sigf(gv1[3][uu]);
        const float c  = fv * c1[idx] + iv * gg;
        c1[idx] = c;
        const float h = ov * tanhf(c);
        h1w[idx] = h;
        hs_bf[(size_t)(t - 1) * kH + idx] = f2bf(h);
      }
    }
    grid_barrier2(syncv, grp);
  }
}

// ---------------- host launcher ----------------
extern "C" void kernel_launch(void* const* d_in, const int* in_sizes, int n_in,
                              void* d_out, int out_size, void* d_ws, size_t ws_size,
                              hipStream_t stream) {
  (void)in_sizes; (void)n_in; (void)out_size; (void)ws_size;
  const float* inputs = (const float*)d_in[0];
  const float* W_in   = (const float*)d_in[1];
  const float* b_in   = (const float*)d_in[2];
  const float* W_ih0  = (const float*)d_in[3];
  const float* W_hh0  = (const float*)d_in[4];
  const float* b_ih0  = (const float*)d_in[5];
  const float* b_hh0  = (const float*)d_in[6];
  const float* W_ih1  = (const float*)d_in[7];
  const float* W_hh1  = (const float*)d_in[8];
  const float* b_ih1  = (const float*)d_in[9];
  const float* b_hh1  = (const float*)d_in[10];
  const float* W_out  = (const float*)d_in[11];
  const float* b_out  = (const float*)d_in[12];
  float* out = (float*)d_out;

  char* p = (char*)d_ws;
  auto carve = [&](size_t bytes) -> char* {
    char* r = p;
    p += (bytes + 255) & ~(size_t)255;
    return r;
  };
  unsigned short* zbf    = (unsigned short*)carve((size_t)kT * kH * 2);
  unsigned short* wih0bf = (unsigned short*)carve((size_t)kG * kH * 2);
  unsigned short* whh0bf = (unsigned short*)carve((size_t)kG * kH * 2);
  unsigned short* wih1bf = (unsigned short*)carve((size_t)kG * kH * 2);
  unsigned short* whh1bf = (unsigned short*)carve((size_t)kG * kH * 2);
  unsigned short* woutbf = (unsigned short*)carve((size_t)kO * kH * 2);
  float*          pre0   = (float*)carve((size_t)kT * kG * 4);
  unsigned short* hsbf   = (unsigned short*)carve((size_t)kT * kH * 2);
  float*          h0buf  = (float*)carve(2 * kH * 4);
  float*          h1buf  = (float*)carve(2 * kH * 4);
  float*          c0     = (float*)carve(kH * 4);
  float*          c1     = (float*)carve(kH * 4);
  unsigned*       syncv  = (unsigned*)carve(4096);

  const int GH = kG * kH;
  cvt_f32_bf16<<<(GH + 255) / 256, 256, 0, stream>>>(W_ih0, wih0bf, GH);
  cvt_f32_bf16<<<(GH + 255) / 256, 256, 0, stream>>>(W_hh0, whh0bf, GH);
  cvt_f32_bf16<<<(GH + 255) / 256, 256, 0, stream>>>(W_ih1, wih1bf, GH);
  cvt_f32_bf16<<<(GH + 255) / 256, 256, 0, stream>>>(W_hh1, whh1bf, GH);
  cvt_f32_bf16<<<(kO * kH + 255) / 256, 256, 0, stream>>>(W_out, woutbf, kO * kH);

  zero_f32<<<(2 * kH + 255) / 256, 256, 0, stream>>>(h0buf, 2 * kH);
  zero_f32<<<(2 * kH + 255) / 256, 256, 0, stream>>>(h1buf, 2 * kH);
  zero_f32<<<(kH + 255) / 256, 256, 0, stream>>>(c0, kH);
  zero_f32<<<(kH + 255) / 256, 256, 0, stream>>>(c1, kH);
  zero_u32<<<1, 1024, 0, stream>>>(syncv, 1024);

  z_kernel<<<(kT * kH + 255) / 256, 256, 0, stream>>>(inputs, W_in, b_in, zbf);

  gemm_pre0<<<(kT / 16) * (kG / 64) / 8, 256, 0, stream>>>(zbf, wih0bf, b_ih0, b_hh0, pre0);

  lstm_scan<<<kNB, 256, kDynLds, stream>>>(pre0, whh0bf, wih1bf, whh1bf, b_ih1, b_hh1,
                                           h0buf, c0, h1buf, c1, hsbf, syncv);

  gemm_out<<<(kT / 16) / 8, 256, 0, stream>>>(hsbf, woutbf, b_out, out);
}
